// SelfAttention_61048665145416
// MI455X (gfx1250) — compile-verified, matrix operand order checked by reading, not run on hardware
//
#include <hip/hip_runtime.h>

// ---------------------------------------------------------------------------
// Types / helpers
// ---------------------------------------------------------------------------
typedef __attribute__((ext_vector_type(16))) __bf16 v16bf;
typedef __attribute__((ext_vector_type(8)))  float  v8f;

struct __attribute__((aligned(16))) U128 { unsigned x, y, z, w; };
struct __attribute__((aligned(16))) F128 { float x, y, z, w; };
union Frag  { v16bf v; struct { U128 lo, hi; } u; };
union Pack8 { U128 u; __bf16 h[8]; };

__device__ inline __bf16 to_bf16(float f) {
  unsigned u = __builtin_bit_cast(unsigned, f);
  unsigned r = (u + 0x7FFFu + ((u >> 16) & 1u)) >> 16;  // round-to-nearest-even
  return __builtin_bit_cast(__bf16, (unsigned short)r);
}

__device__ inline v8f wmma_bf16(v16bf a, v16bf b, v8f c) {
  // D = A(16x32 bf16) * B(32x16 bf16) + C(16x16 f32)
  return __builtin_amdgcn_wmma_f32_16x16x32_bf16(
      /*neg_a=*/false, a, /*neg_b=*/false, b,
      /*c_mod=*/(short)0, c, /*reuse_a=*/false, /*reuse_b=*/false);
}

// Load a 16x32 bf16 A-fragment (or its mirrored B-fragment) from a row-major
// LDS tile. ISA layout: lane covers row (lane&15); K chunks are
// kb..kb+7 and kb+16..kb+23 with kb = 8*(lane>=16). Two ds_load_b128.
__device__ inline v16bf ld_frag(const __bf16* tile, int row0, int stride, int k0) {
  const int lane = threadIdx.x & 31;
  const __bf16* p = tile + (size_t)(row0 + (lane & 15)) * stride
                         + k0 + ((lane >> 4) << 3);
  Frag f;
  f.u.lo = *(const U128*)p;
  f.u.hi = *(const U128*)(p + 16);
  return f.v;
}

// Stage 16 contiguous elements into LDS as bf16 (f32 source converts).
__device__ inline void cp16(const float* s, __bf16* d) {
#pragma unroll
  for (int i = 0; i < 16; i += 4) {
    F128 v = *(const F128*)(s + i);
    d[i + 0] = to_bf16(v.x); d[i + 1] = to_bf16(v.y);
    d[i + 2] = to_bf16(v.z); d[i + 3] = to_bf16(v.w);
  }
}
__device__ inline void cp16(const __bf16* s, __bf16* d) {
  *(U128*)d       = *(const U128*)s;
  *(U128*)(d + 8) = *(const U128*)(s + 8);
}

__device__ inline void st_out(float*  p, float v) { *p = v; }
__device__ inline void st_out(__bf16* p, float v) { *p = to_bf16(v); }

// ---------------------------------------------------------------------------
// Tensor Data Mover: 2D bf16 tile, Global -> LDS, with LDS row padding.
// D# per CDNA5 ISA ch.8: group0 {count=1, lds_addr, global_addr, type=2},
// group1 {data_size=2B, pad_enable, pad_interval/amount, dims, stride}.
// Issued from one wave (EXEC ignored); completion via TENSORcnt.
// This toolchain exposes the 6-arg builtin:
//   (uint32x4 g0, int32x8 g1, int32x4 g2, int32x4 g3, int32x8, i32 cpol)
// ---------------------------------------------------------------------------
#if defined(__AMDGCN__) && __has_builtin(__builtin_amdgcn_tensor_load_to_lds) && \
    __has_builtin(__builtin_amdgcn_s_wait_tensorcnt)
#define HAVE_TDM 1
typedef unsigned tdm_u32x4 __attribute__((ext_vector_type(4)));
typedef int      tdm_i32x8 __attribute__((ext_vector_type(8)));
typedef int      tdm_i32x4 __attribute__((ext_vector_type(4)));

// tile_w bf16 elems per row, `rows` rows, global row stride g_stride elems.
// After every (2<<pi_code) DWORDs written to LDS, skip (pa_code+1) DWORDs.
__device__ inline void tdm_load_tile(const void* g, unsigned lds_byte,
                                     unsigned tile_w, unsigned rows,
                                     unsigned g_stride, unsigned pi_code,
                                     unsigned pa_code) {
  const unsigned long long ga = (unsigned long long)g;
  tdm_u32x4 g0;
  g0[0] = 1u;                                              // count=1 (valid D#)
  g0[1] = lds_byte;                                        // lds_addr
  g0[2] = (unsigned)ga;                                    // global_addr[31:0]
  g0[3] = (unsigned)((ga >> 32) & 0x1FFFFFFu) | (2u << 30);// ga[56:32] | type=2
  tdm_i32x8 g1;
  g1[0] = (int)((1u << 16)            // data_size = 2 bytes
              | (1u << 20)            // pad_enable
              | (pi_code << 22)       // pad_interval
              | (pa_code << 25));     // pad_amount
  g1[1] = (int)(tile_w << 16);                      // tensor_dim0[15:0]
  g1[2] = (int)((tile_w >> 16) | (rows << 16));     // tdim0[31:16]|tdim1[15:0]
  g1[3] = (int)((rows >> 16) | (tile_w << 16));     // tdim1[31:16]|tile_dim0
  g1[4] = (int)rows;                                // tile_dim1 (tile_dim2=0)
  g1[5] = (int)g_stride;                            // tensor_dim0_stride[31:0]
  g1[6] = 0;
  g1[7] = 0;
  tdm_i32x4 z4 = {0, 0, 0, 0};                      // D# groups 2/3 (2D tensor)
  tdm_i32x8 z8 = {0, 0, 0, 0, 0, 0, 0, 0};
  __builtin_amdgcn_tensor_load_to_lds(g0, g1, z4, z4, z8, 0);
}
#endif

// ---------------------------------------------------------------------------
// GEMM (NT): C[M,N] = A[M,K] * B[N,K]^T + bias[N]
// Block tile 128x128x32, 256 threads = 8 waves (2 M x 4 N), wave tile 64x32.
// ---------------------------------------------------------------------------
template <typename AT, typename OT>
__global__ __launch_bounds__(256) void gemm_nt_bias(
    const AT* __restrict__ A, const float* __restrict__ Bw,
    const float* __restrict__ bias, OT* __restrict__ C,
    int M, int N, int K) {
  __shared__ __bf16 As[128 * 40];   // 128 rows x 32 K, stride 40 (pad)
  __shared__ __bf16 Bs[128 * 40];

  const int tid  = threadIdx.x;
  const int lane = tid & 31, wave = tid >> 5;
  const int wm = wave & 1, wn = wave >> 1;
  const int m0 = blockIdx.y * 128, n0 = blockIdx.x * 128;
  const int sr = tid >> 1, sc = (tid & 1) << 4;   // staging: row, col16

  v8f acc[4][2] = {};

  for (int k0 = 0; k0 < K; k0 += 32) {
    cp16(&A [(size_t)(m0 + sr) * K + k0 + sc], &As[sr * 40 + sc]);
    cp16(&Bw[(size_t)(n0 + sr) * K + k0 + sc], &Bs[sr * 40 + sc]);
    if (k0 + 32 < K) {  // prefetch next K slice into GL2 while we compute
      __builtin_prefetch(&A [(size_t)(m0 + sr) * K + k0 + 32 + sc], 0, 1);
      __builtin_prefetch(&Bw[(size_t)(n0 + sr) * K + k0 + 32 + sc], 0, 1);
    }
    __syncthreads();

    v16bf af[4], bfrag[2];
#pragma unroll
    for (int mi = 0; mi < 4; ++mi) af[mi]    = ld_frag(As, wm * 64 + mi * 16, 40, 0);
#pragma unroll
    for (int ni = 0; ni < 2; ++ni) bfrag[ni] = ld_frag(Bs, wn * 32 + ni * 16, 40, 0);
#pragma unroll
    for (int mi = 0; mi < 4; ++mi)
#pragma unroll
      for (int ni = 0; ni < 2; ++ni)
        acc[mi][ni] = wmma_bf16(af[mi], bfrag[ni], acc[mi][ni]);
    __syncthreads();
  }

  // Epilogue: C layout = lane covers col (lane&15), rows r + 8*(lane>=16).
  const int hr = (lane >> 4) << 3, n = lane & 15;
#pragma unroll
  for (int ni = 0; ni < 2; ++ni) {
    const int col = n0 + wn * 32 + ni * 16 + n;
    const float bv = bias[col];
#pragma unroll
    for (int mi = 0; mi < 4; ++mi)
#pragma unroll
      for (int r = 0; r < 8; ++r) {
        const int row = m0 + wm * 64 + mi * 16 + hr + r;
        st_out(C + (size_t)row * N + col, acc[mi][ni][r] + bv);
      }
  }
}

// ---------------------------------------------------------------------------
// Fused flash attention over full d_embed=512, scale 1/sqrt(64).
// Grid: (S/64, B). Block: 256 threads = 8 waves.
// Wave w: query rows (w>>1)*16..+15 of the 64-row tile; V columns (w&1)*256..+255.
// K-tiles of 32 keys; online softmax; P routed through LDS for layout change.
// Q/K tiles staged by the TDM (row pad 4 DWORDs per 256 -> LDS stride 520).
// ---------------------------------------------------------------------------
__global__ __launch_bounds__(256) void flash_attn(
    const __bf16* __restrict__ qkv,   // [B*S, 1536] : q|k|v
    __bf16* __restrict__ attn) {      // [B*S, 512]
  extern __shared__ __bf16 smem[];
  __bf16* Qs = smem;                  // 64  x 520   (LDS byte off 0)
  __bf16* Ks = Qs + 64 * 520;         // 32  x 520   (LDS byte off 66560)
  __bf16* Vt = Ks + 32 * 520;         // 512 x 40    (transposed: [d][key])
  __bf16* Ps = Vt + 512 * 40;         // 64  x 40

  const int tid  = threadIdx.x;
  const int lane = tid & 31, wave = tid >> 5;
  const int qsub = wave >> 1;         // which 16 query rows
  const int half = wave & 1;          // which 256 output columns
  const size_t qrow0 = (size_t)blockIdx.y * 4096 + (size_t)blockIdx.x * 64;

  // Stage Q tile once (already bf16 in workspace).
#ifdef HAVE_TDM
  if (wave == 0)
    tdm_load_tile(&qkv[qrow0 * 1536], 0u, 512u, 64u, 1536u, 7u, 3u);
#else
  for (int cid = tid; cid < 4096; cid += 256) {
    const int row = cid >> 6, c8 = (cid & 63) << 3;
    *(U128*)&Qs[row * 520 + c8] = *(const U128*)&qkv[(qrow0 + row) * 1536 + c8];
  }
#endif

  float mrow[8], lrow[8];
  v8f o[16] = {};
#pragma unroll
  for (int r = 0; r < 8; ++r) { mrow[r] = -1e30f; lrow[r] = 0.0f; }
  const float scale = 0.125f;         // 1/sqrt(D_HEAD=64)

  for (int kt = 0; kt < 128; ++kt) {
    const size_t krow0 = (size_t)blockIdx.y * 4096 + (size_t)kt * 32;
    __syncthreads();                  // prev PV done before restaging
#ifdef HAVE_TDM
    if (wave == 0)
      tdm_load_tile(&qkv[krow0 * 1536 + 512], 66560u, 512u, 32u, 1536u, 7u, 3u);
#else
    for (int cid = tid; cid < 2048; cid += 256) {
      const int row = cid >> 6, c8 = (cid & 63) << 3;
      *(U128*)&Ks[row * 520 + c8] =
          *(const U128*)&qkv[(krow0 + row) * 1536 + 512 + c8];
    }
#endif
    // V tile: transpose through VGPRs into Vt[d][key] (TDM cannot transpose).
    for (int cid = tid; cid < 2048; cid += 256) {
      const int row = cid >> 6, c8 = (cid & 63) << 3;
      Pack8 p; p.u = *(const U128*)&qkv[(krow0 + row) * 1536 + 1024 + c8];
#pragma unroll
      for (int j = 0; j < 8; ++j) Vt[(c8 + j) * 40 + row] = p.h[j];
    }
    // Prefetch next K/V tile into GL2 behind this tile's 48 WMMAs.
    if (kt + 1 < 128) {
      const size_t nrow0 = (size_t)blockIdx.y * 4096 + (size_t)(kt + 1) * 32;
      const int prow = (tid & 127) >> 2;
      const int pcol = ((tid >> 7) ? 1024 : 512) + ((tid & 3) << 7);
      __builtin_prefetch(&qkv[(nrow0 + prow) * 1536 + pcol], 0, 1);
    }
#ifdef HAVE_TDM
    __builtin_amdgcn_s_wait_tensorcnt(0);   // waves w/o TDM ops pass trivially
#endif
    __syncthreads();

    // S = Q_sub(16x512) . K_tile(32x512)^T
    v8f s0 = {}, s1 = {};
#pragma unroll 4
    for (int kk = 0; kk < 16; ++kk) {
      v16bf aq  = ld_frag(Qs, qsub * 16, 520, kk * 32);
      v16bf k0f = ld_frag(Ks, 0,  520, kk * 32);
      v16bf k1f = ld_frag(Ks, 16, 520, kk * 32);
      s0 = wmma_bf16(aq, k0f, s0);
      s1 = wmma_bf16(aq, k1f, s1);
    }

    // Online softmax: row stats via 16-lane shfl reductions (wave32 halves).
    float corr[8];
#pragma unroll
    for (int r = 0; r < 8; ++r) {
      float a = s0[r] * scale, c = s1[r] * scale;
      float t = fmaxf(a, c);
#pragma unroll
      for (int off = 1; off < 16; off <<= 1) t = fmaxf(t, __shfl_xor(t, off, 32));
      const float mn = fmaxf(mrow[r], t);
      const float cf = __expf(mrow[r] - mn);
      const float p0 = __expf(a - mn), p1 = __expf(c - mn);
      float rs = p0 + p1;
#pragma unroll
      for (int off = 1; off < 16; off <<= 1) rs += __shfl_xor(rs, off, 32);
      lrow[r] = lrow[r] * cf + rs;
      mrow[r] = mn;
      corr[r] = cf;
      s0[r] = p0; s1[r] = p1;
    }
#pragma unroll
    for (int ni = 0; ni < 16; ++ni)
#pragma unroll
      for (int r = 0; r < 8; ++r) o[ni][r] *= corr[r];

    // Route P (D-layout) through LDS into A-fragment layout.
    if (half == 0) {
      const int hr = (lane >> 4) << 3, n = lane & 15;
#pragma unroll
      for (int r = 0; r < 8; ++r) {
        Ps[(qsub * 16 + hr + r) * 40 + n]      = to_bf16(s0[r]);
        Ps[(qsub * 16 + hr + r) * 40 + 16 + n] = to_bf16(s1[r]);
      }
    }
    __syncthreads();

    // O += P(16x32) . V_tile(32x512) on this wave's 256 columns.
    v16bf ap = ld_frag(Ps, qsub * 16, 40, 0);
#pragma unroll
    for (int ni = 0; ni < 16; ++ni) {
      v16bf bv = ld_frag(Vt, half * 256 + ni * 16, 40, 0);
      o[ni] = wmma_bf16(ap, bv, o[ni]);
    }
  }

  // Normalize and store as bf16 for the output projection.
  const int hr = (lane >> 4) << 3, n = lane & 15;
#pragma unroll
  for (int r = 0; r < 8; ++r) lrow[r] = 1.0f / lrow[r];
#pragma unroll
  for (int ni = 0; ni < 16; ++ni)
#pragma unroll
    for (int r = 0; r < 8; ++r) {
      const size_t row = qrow0 + qsub * 16 + hr + r;
      attn[row * 512 + half * 256 + ni * 16 + n] = to_bf16(o[ni][r] * lrow[r]);
    }
}

// ---------------------------------------------------------------------------
// Launch: qkv-proj GEMM -> fused flash attention -> out-proj GEMM
// ---------------------------------------------------------------------------
extern "C" void kernel_launch(void* const* d_in, const int* in_sizes, int n_in,
                              void* d_out, int out_size, void* d_ws, size_t ws_size,
                              hipStream_t stream) {
  (void)in_sizes; (void)n_in; (void)out_size; (void)ws_size;
  const float* x     = (const float*)d_in[0];  // [4,4096,512]
  const float* w_in  = (const float*)d_in[1];  // [1536,512]
  const float* b_in  = (const float*)d_in[2];  // [1536]
  const float* w_out = (const float*)d_in[3];  // [512,512]
  const float* b_out = (const float*)d_in[4];  // [512]
  float* out = (float*)d_out;                  // [4,4096,512] f32

  __bf16* qkv  = (__bf16*)d_ws;                                      // 16384x1536
  __bf16* attn = (__bf16*)((char*)d_ws + (size_t)16384 * 1536 * 2);  // 16384x512

  const dim3 blk(256);

  // 1) qkv = x @ w_in^T + b_in  (f32 in, bf16 out)
  gemm_nt_bias<float, __bf16><<<dim3(1536 / 128, 16384 / 128), blk, 0, stream>>>(
      x, w_in, b_in, qkv, 16384, 1536, 512);

  // 2) fused attention (softmax(q k^T / 8) v), bf16 out
  const size_t lds_bytes =
      (size_t)(64 * 520 + 32 * 520 + 512 * 40 + 64 * 40) * sizeof(__bf16); // 145920
  flash_attn<<<dim3(4096 / 64, 4), blk, lds_bytes, stream>>>(qkv, attn);

  // 3) out = attn @ w_out^T + b_out  (bf16 in, f32 out)
  gemm_nt_bias<__bf16, float><<<dim3(512 / 128, 16384 / 128), blk, 0, stream>>>(
      attn, w_out, b_out, out, 16384, 512, 512);
}
